// STDPLearner_70085276336702
// MI455X (gfx1250) — matrix operand order repaired and checked by reading, not compiled, not run on hardware
//
#include <hip/hip_runtime.h>
#include <hip/hip_bf16.h>

typedef __attribute__((ext_vector_type(16))) __bf16 v16bf;
typedef __attribute__((ext_vector_type(8)))  float  v8f;

#define B_    16
#define CIN   64
#define H_    66
#define W_    66
#define COUT  128
#define HOUT  64
#define WOUT  64

#define N_PRE  (B_*CIN*H_*W_)        // 4460544
#define N_POST (B_*COUT*HOUT*WOUT)   // 8388608
#define N_DW   (COUT*CIN*9)          // 73728
#define WS_FLOATS (2*9*COUT*CIN)     // 147456

#define ROWS_PER_CHUNK 16
#define NCHUNK ((B_*HOUT)/ROWS_PER_CHUNK)  // 64

#define BSTRIDE 20   // uints per shifted B row (16 used); mult of 4 -> 16B align, pad kills bank conflicts

union BPack { unsigned int u; __bf16 b[2]; };
union Frag  { unsigned int u[8]; uint4 q[2]; v16bf v; };

__device__ __forceinline__ unsigned int packf(float a, float b) {
    BPack p; p.b[0] = (__bf16)a; p.b[1] = (__bf16)b; return p.u;  // v_cvt_pk_bf16_f32
}

// ---------------------------------------------------------------- kernels
__global__ void zero_ws_kernel(float* __restrict__ ws) {
    int i = blockIdx.x * blockDim.x + threadIdx.x;
    if (i < WS_FLOATS) ws[i] = 0.0f;
}

__global__ void trace_pre_kernel(const float* __restrict__ trace_pre,
                                 const float* __restrict__ in_spike,
                                 float* __restrict__ out) {
    int i = blockIdx.x * blockDim.x + threadIdx.x;
    if (i < N_PRE) out[i] = 0.5f * trace_pre[i] + in_spike[i];
}

__global__ void trace_post_kernel(const float* __restrict__ trace_post,
                                  const float* __restrict__ out_spike,
                                  float* __restrict__ out) {
    int i = blockIdx.x * blockDim.x + threadIdx.x;
    if (i < N_POST) out[i] = 0.5f * trace_post[i] + out_spike[i];
}

// Split-K STDP correlation GEMM.
//   g=0 : s_post[o,i] = sum_{b,p,q} out_spike[b,o,p,q] * tr_pre[b,i,p+kh,q+kw]
//   g=1 : s_pre [o,i] = sum_{b,p,q} tr_post [b,o,p,q] * in_spike[b,i,p+kh,q+kw]
// grid = (NCHUNK, 3 /*kh*/), block = 512 (16 waves). Wave w: g=w>>3, mblk=w&7.
__global__ __launch_bounds__(512, 1)
void stdp_gemm_kernel(const float* __restrict__ in_spike,
                      const float* __restrict__ out_spike,
                      const float* __restrict__ tr_pre,
                      const float* __restrict__ tr_post,
                      float* __restrict__ ws) {
    // B pre-split to bf16 AND pre-shifted per kw: [g][kw][plane hi/lo][row][uint]
    // B fragment window for (kw,hh) is then a 16B-aligned 32-byte direct load.
    __shared__ unsigned int sB[2][3][2][CIN][BSTRIDE];   // 61440 B

    const int tid  = threadIdx.x;
    const int lane = tid & 31;
    const int wv   = tid >> 5;        // 0..15
    const int g    = wv >> 3;         // 0 or 1
    const int mblk = wv & 7;          // M block (16 rows of o)
    const int kh   = blockIdx.y;      // 0..2

    const int hh = lane >> 4;         // lane half
    const int lm = lane & 15;

    // staging roles (block-wide): 4 threads per (g,row)
    const int sg = tid >> 8;          // 0..1 : which gemm's B to stage
    const int si = (tid & 255) >> 2;  // 0..63: CIN row
    const int sc = tid & 3;           // uint column phase

    // per-wave A source (one wave owns 16 consecutive o-rows; lane -> one row)
    const float* Abase = (g ? tr_post : out_spike)
                       + (size_t)(mblk * 16 + lm) * (HOUT * WOUT);

    v8f acc[3][4];
    #pragma unroll
    for (int k = 0; k < 3; ++k)
        #pragma unroll
        for (int n = 0; n < 4; ++n)
            #pragma unroll
            for (int v = 0; v < 8; ++v)
                acc[k][n][v] = 0.0f;

    for (int r = 0; r < ROWS_PER_CHUNK; ++r) {
        const int row = blockIdx.x * ROWS_PER_CHUNK + r;
        const int b = row >> 6;       // batch
        const int p = row & 63;       // output row

        const float* Arow = Abase + (size_t)b * (COUT * HOUT * WOUT) + p * WOUT;
        const float* Bsrc = (sg ? in_spike : tr_pre)
                          + (size_t)((b * CIN + si) * H_ + (p + kh)) * W_;

        #pragma unroll
        for (int kstep = 0; kstep < 2; ++kstep) {
            const int q0 = kstep * 32;
            __syncthreads();

            // --- stage B: hi/lo split + 3 kw-shifted packed copies, once per block
            {
                const float* src = Bsrc + q0;
                #pragma unroll
                for (int u = sc; u < 16; u += 4) {
                    float v0 = src[2 * u];
                    float v1 = src[2 * u + 1];
                    float v2 = src[2 * u + 2];
                    float v3 = src[2 * u + 3];
                    float l0 = v0 - (float)(__bf16)v0;
                    float l1 = v1 - (float)(__bf16)v1;
                    float l2 = v2 - (float)(__bf16)v2;
                    float l3 = v3 - (float)(__bf16)v3;
                    sB[sg][0][0][si][u] = packf(v0, v1);
                    sB[sg][1][0][si][u] = packf(v1, v2);
                    sB[sg][2][0][si][u] = packf(v2, v3);
                    sB[sg][0][1][si][u] = packf(l0, l1);
                    sB[sg][1][1][si][u] = packf(l1, l2);
                    sB[sg][2][1][si][u] = packf(l2, l3);
                }
                __builtin_prefetch(src + W_, 0, 1);   // next p-row
            }
            __syncthreads();

            // --- A fragment: direct global load (2 x 32B runs), split in regs
            v16bf a_hi, a_lo;
            {
                const float4* r0 = (const float4*)(Arow + q0 + hh * 8);
                const float4* r1 = (const float4*)(Arow + q0 + 16 + hh * 8);
                float a[16];
                float4 x0 = r0[0], x1 = r0[1], y0 = r1[0], y1 = r1[1];
                a[0]=x0.x; a[1]=x0.y; a[2]=x0.z; a[3]=x0.w;
                a[4]=x1.x; a[5]=x1.y; a[6]=x1.z; a[7]=x1.w;
                a[8]=y0.x; a[9]=y0.y; a[10]=y0.z; a[11]=y0.w;
                a[12]=y1.x; a[13]=y1.y; a[14]=y1.z; a[15]=y1.w;
                #pragma unroll
                for (int j = 0; j < 16; ++j) {
                    __bf16 h = (__bf16)a[j];
                    a_hi[j] = h;
                    a_lo[j] = (__bf16)(a[j] - (float)h);
                }
                __builtin_prefetch(Arow + q0 + WOUT, 0, 1);   // next p-row
            }

            #pragma unroll
            for (int nb = 0; nb < 4; ++nb) {
                const int brow = nb * 16 + lm;
                #pragma unroll
                for (int kw = 0; kw < 3; ++kw) {
                    // fragments are direct aligned 32B LDS loads (2x ds_load_b128 each)
                    Frag bh, bl;
                    const uint4* ph = (const uint4*)&sB[g][kw][0][brow][hh * 4];
                    const uint4* pl = (const uint4*)&sB[g][kw][1][brow][hh * 4];
                    bh.q[0] = ph[0]; bh.q[1] = ph[2];
                    bl.q[0] = pl[0]; bl.q[1] = pl[2];
                    // 3-term bf16 split product, f32 accumulate (~f32 accuracy)
                    acc[kw][nb] = __builtin_amdgcn_wmma_f32_16x16x32_bf16(
                        false, a_hi, false, bh.v, (short)0, acc[kw][nb], false, false);
                    acc[kw][nb] = __builtin_amdgcn_wmma_f32_16x16x32_bf16(
                        false, a_hi, false, bl.v, (short)0, acc[kw][nb], false, false);
                    acc[kw][nb] = __builtin_amdgcn_wmma_f32_16x16x32_bf16(
                        false, a_lo, false, bh.v, (short)0, acc[kw][nb], false, false);
                }
            }
        }
    }

    // --- accumulate partials: ws[g][kh*3+kw][o][i]
    #pragma unroll
    for (int kw = 0; kw < 3; ++kw) {
        const int off = kh * 3 + kw;
        #pragma unroll
        for (int nb = 0; nb < 4; ++nb) {
            const int N = nb * 16 + lm;
            #pragma unroll
            for (int v = 0; v < 8; ++v) {
                const int M = mblk * 16 + v + hh * 8;   // C/D layout: VGPR v -> M=v / v+8
                atomicAdd(&ws[((size_t)(g * 9 + off) * COUT + M) * CIN + N],
                          acc[kw][nb][v]);
            }
        }
    }
}

__global__ void finalize_kernel(const float* __restrict__ weight,
                                const float* __restrict__ ws,
                                float* __restrict__ dw) {
    int idx = blockIdx.x * blockDim.x + threadIdx.x;
    if (idx >= N_DW) return;
    // weight/dw layout: [o][i][kh][kw] flat -> o*576 + i*9 + k
    int o = idx / (CIN * 9);
    int rem = idx % (CIN * 9);
    int i = rem / 9;
    int k = rem % 9;
    float s_post = ws[((size_t)(0 * 9 + k) * COUT + o) * CIN + i];
    float s_pre  = ws[((size_t)(1 * 9 + k) * COUT + o) * CIN + i];
    dw[idx] = weight[idx] * (s_post - s_pre);
}

// ---------------------------------------------------------------- launcher
extern "C" void kernel_launch(void* const* d_in, const int* in_sizes, int n_in,
                              void* d_out, int out_size, void* d_ws, size_t ws_size,
                              hipStream_t stream) {
    (void)in_sizes; (void)n_in; (void)out_size; (void)ws_size;
    const float* in_spike   = (const float*)d_in[0];
    const float* out_spike  = (const float*)d_in[1];
    const float* trace_pre  = (const float*)d_in[2];
    const float* trace_post = (const float*)d_in[3];
    const float* weight     = (const float*)d_in[4];

    float* out         = (float*)d_out;
    float* tr_pre_out  = out;
    float* tr_post_out = out + N_PRE;
    float* dw_out      = out + N_PRE + N_POST;
    float* ws          = (float*)d_ws;

    zero_ws_kernel<<<(WS_FLOATS + 255) / 256, 256, 0, stream>>>(ws);
    trace_pre_kernel<<<(N_PRE + 255) / 256, 256, 0, stream>>>(trace_pre, in_spike, tr_pre_out);
    trace_post_kernel<<<(N_POST + 255) / 256, 256, 0, stream>>>(trace_post, out_spike, tr_post_out);

    dim3 grid(NCHUNK, 3, 1);
    stdp_gemm_kernel<<<grid, 512, 0, stream>>>(in_spike, out_spike,
                                               tr_pre_out, tr_post_out, ws);

    finalize_kernel<<<(N_DW + 255) / 256, 256, 0, stream>>>(weight, ws, dw_out);
}
